// DISPATCHLoss_25305947308060
// MI455X (gfx1250) — compile-verified, matrix operand order checked by reading, not run on hardware
//
#include <hip/hip_runtime.h>
#include <hip/hip_bf16.h>

// ---------------------------------------------------------------------------
// STFT-mag patch distillation loss for MI455X (gfx1250).
//
// rFFT (no window) == two dense GEMMs vs fixed cos/sin f16 bases:
// ~108 GFLOP f16 WMMA (compute-bound, ~20us), ~230 MB HBM (~10us @23.3TB/s).
// Each wave owns one 16x16 patch tile for all 3 signals (6 f32 accumulators),
// so per-patch stats reduce fully in-wave, no atomics.
//
// Round-3 state: 99KB LDS with de-overlapped frame rows (stride 1032 halves ->
// lane dword stride 516 == 4 mod 64 banks, conflict-free A-fragment loads)
// compiled clean: no spills, clause'd loads, split-counter waits.
// Round-4: unroll k-loop x4 (was x2) + per-signal row pointers to (a) amortize
// loop control, (b) stop the allocator from recycling in-flight WMMA operand
// registers for the third signal's LDS address adds, which forced two v_nop x4
// WMMA->VALU WAR hazard bursts per iteration (~10% issue tax).
// ---------------------------------------------------------------------------

typedef _Float16 v8h  __attribute__((ext_vector_type(8)));
typedef _Float16 v16h __attribute__((ext_vector_type(16)));
typedef float    v8f  __attribute__((ext_vector_type(8)));

constexpr int BATCH  = 16;
constexpr int L      = 262144;
constexpr int NFFT   = 1024;
constexpr int HOP    = 256;
constexpr int PS     = 16;
constexpr int NFREQ  = NFFT / 2 + 1;          // 513
constexpr int NTIME  = 1 + L / HOP;           // 1025
constexpr int NF     = (NFREQ + PS - 1) / PS; // 33 freq patches (pad to 528)
constexpr int NT     = (NTIME + PS - 1) / PS; // 65 time patches (pad to 1040)
constexpr int NPATCH = NF * NT;               // 2145
constexpr int KSEL   = 643;                   // int(0.3 * 2145)
constexpr int KCH    = NFFT / 32;             // 32 k-chunks of 32 (WMMA K)
constexpr int SROW   = NFFT + 8;              // 1032: LDS frame-row stride
constexpr int BASIS_HALVES = NF * KCH * 32 * 16; // 540672 halves per basis
constexpr float EPSV = 1e-8f;

// ---------------------------------------------------------------------------
// Kernel 0: one-time f32 -> f16 conversion of all 3 waveforms into ws.
// ---------------------------------------------------------------------------
__global__ void to_half(const float* __restrict__ xs, const float* __restrict__ xt,
                        const float* __restrict__ xg, _Float16* __restrict__ xh) {
  size_t t = (size_t)blockIdx.x * blockDim.x + threadIdx.x; // one per 8 samples
  size_t n8 = (size_t)BATCH * L / 8;
  if (t >= 3 * n8) return;
  int sig = (int)(t / n8);
  size_t off = (t - (size_t)sig * n8) * 8;
  const float* src = (sig == 0) ? xs : ((sig == 1) ? xt : xg);
  v8f v = *(const v8f*)(src + off);
  v8h h;
#pragma unroll
  for (int i = 0; i < 8; ++i) h[i] = (_Float16)v[i];
  *(v8h*)(xh + (size_t)sig * BATCH * L + off) = h;
}

// ---------------------------------------------------------------------------
// Kernel 1: cos/sin DFT bases in f16, pre-swizzled into the per-lane WMMA
// B-fragment layout: flat = ((nt*KCH + kc)*32 + lane)*16 + j, with
//   kbase = lane<16 ? 0 : 8;  k = kc*32 + kbase + (j<8 ? j : 8+j)
//   n     = nt*16 + (lane & 15)
// Bins n >= 513 zeroed (matches jnp.pad-after-EPS-clamp).
// ---------------------------------------------------------------------------
__global__ void build_basis(_Float16* __restrict__ bre, _Float16* __restrict__ bim) {
  int tid = blockIdx.x * blockDim.x + threadIdx.x;
  if (tid >= BASIS_HALVES) return;
  int j    = tid & 15;
  int lane = (tid >> 4) & 31;
  int kc   = (tid >> 9) & 31;
  int nt   = tid >> 14;
  int kbase = (lane < 16) ? 0 : 8;
  int k = kc * 32 + kbase + ((j < 8) ? j : (8 + j)); // j>=8 -> 16+(j-8)
  int n = nt * 16 + (lane & 15);
  float re = 0.0f, im = 0.0f;
  if (n < NFREQ) {
    int m = (k * n) & (NFFT - 1);                // phase mod 2*pi
    float a = (float)m * 6.135923151542565e-3f;  // 2*pi/1024
    sincosf(a, &im, &re);                        // sign of sin irrelevant (|.|)
  }
  bre[tid] = (_Float16)re;
  bim[tid] = (_Float16)im;
}

__device__ __forceinline__ v8f wmma16(v16h a, v16h b, v8f c) {
  return __builtin_amdgcn_wmma_f32_16x16x32_f16(
      /*neg_a=*/false, a, /*neg_b=*/false, b,
      /*c_mod=*/(short)0, c, /*reuse_a=*/false, /*reuse_b=*/false);
}

// A-fragment (16x32 f16) per ISA layout: lanes 0-15 hold M=lane,
// K = {0..7} U {16..23}; lanes 16-31 hold M=lane-16, K = {8..15} U {24..31}.
// Both 8-half runs are 16B aligned in LDS -> two ds_load_b128.
__device__ __forceinline__ v16h load_a(const _Float16* w, int off) {
  v8h lo = *(const v8h*)(w + off);
  v8h hi = *(const v8h*)(w + off + 16);
  v16h a;
#pragma unroll
  for (int i = 0; i < 8; ++i) { a[i] = lo[i]; a[8 + i] = hi[i]; }
  return a;
}

// ---------------------------------------------------------------------------
// Kernel 2: fused STFT-GEMM + patch stats.
// grid = (NT time-patches, BATCH), block = 256 (8 waves).
// Wave w handles freq-patch tiles nt = w, w+8, ...
// ---------------------------------------------------------------------------
__global__ __launch_bounds__(256)
void stft_patch_stats(const _Float16* __restrict__ xh,
                      const _Float16* __restrict__ bre, const _Float16* __restrict__ bim,
                      float* __restrict__ errS, float* __restrict__ errT,
                      float* __restrict__ pl) {
  // De-overlapped frame rows, stride SROW halves: lane dword stride 516 == 4
  // (mod 64) -> A-fragment ds_load_b128 hits 16 distinct bank groups.
  __shared__ _Float16 aw[3][PS * SROW]; // 3 * 16 * 1032 * 2B = 99072 B

  const int tpatch = blockIdx.x;
  const int b      = blockIdx.y;
  const int base   = tpatch * PS * HOP;

  // Fill: 8-sample chunks, vectorized b128 path with scalar reflect fallback.
  for (int c = threadIdx.x; c < 3 * PS * (NFFT / 8); c += 256) {
    int sig = c >> 11;              // / (16*128)
    int r   = c & 2047;
    int m   = r >> 7;               // frame row
    int k0  = (r & 127) << 3;       // sample-in-frame
    int v0  = base + m * HOP + k0;  // index into virtual reflect-padded signal
    const _Float16* src = xh + (size_t)sig * BATCH * L + (size_t)b * L;
    _Float16* dst = &aw[sig][m * SROW + k0];
    if (v0 >= NFFT / 2 && v0 + 8 <= L + NFFT / 2) {
      *(v8h*)dst = *(const v8h*)(src + (v0 - NFFT / 2));
    } else {
#pragma unroll
      for (int j = 0; j < 8; ++j) {
        int v = v0 + j;
        int g = (v < NFFT / 2)      ? (NFFT / 2 - v)
              : (v >= L + NFFT / 2) ? (2 * L + NFFT / 2 - 2 - v)
                                    : (v - NFFT / 2);
        dst[j] = src[g];
      }
    }
  }
  __syncthreads();

  const int wid   = threadIdx.x >> 5;
  const int lane  = threadIdx.x & 31;
  const int mrow  = lane & 15;
  const int kbase = (lane < 16) ? 0 : 8;
  const int mhi   = (lane >> 4) << 3; // 0 or 8: C/D VGPR row offset
  const int abase = mrow * SROW + kbase;

  // Per-signal row pointers: keeps LDS address math off in-flight WMMA regs.
  const _Float16* __restrict__ w0 = &aw[0][abase];
  const _Float16* __restrict__ w1 = &aw[1][abase];
  const _Float16* __restrict__ w2 = &aw[2][abase];

  for (int nt = wid; nt < NF; nt += 8) {
    v8f rs = {}, is_ = {}, rt = {}, it = {}, rg = {}, ig = {};
    const v16h* pre = (const v16h*)(bre + (size_t)nt * (KCH * 512));
    const v16h* pim = (const v16h*)(bim + (size_t)nt * (KCH * 512));

    // Bounded unroll: x4 amortizes loop control without spilling (x32 spills).
#pragma unroll 4
    for (int kc = 0; kc < KCH; ++kc) {
      const int ko = kc * 32;
      v16h aS = load_a(w0, ko);
      v16h aT = load_a(w1, ko);
      v16h aG = load_a(w2, ko);
      v16h fre = pre[kc * 32 + lane]; // one 32B load, L2-resident basis
      v16h fim = pim[kc * 32 + lane];
      rs = wmma16(aS, fre, rs);  is_ = wmma16(aS, fim, is_);
      rt = wmma16(aT, fre, rt);  it  = wmma16(aT, fim, it);
      rg = wmma16(aG, fre, rg);  ig  = wmma16(aG, fim, ig);
    }

    // Epilogue: magnitudes + EPS clamp on valid region, per-patch stats.
    const int fcol = nt * 16 + (lane & 15);
    float sa = 0.f, sb = 0.f, sc = 0.f;
#pragma unroll
    for (int r = 0; r < 8; ++r) {
      int tg  = tpatch * 16 + r + mhi;
      bool ok = (fcol < NFREQ) && (tg < NTIME);
      float ms = sqrtf(rs[r] * rs[r] + is_[r] * is_[r]);
      float mt = sqrtf(rt[r] * rt[r] + it[r]  * it[r]);
      float mg = sqrtf(rg[r] * rg[r] + ig[r]  * ig[r]);
      ms = ok ? fmaxf(ms, EPSV) : 0.f;
      mt = ok ? fmaxf(mt, EPSV) : 0.f;
      mg = ok ? fmaxf(mg, EPSV) : 0.f;
      sa += fabsf(ms - mg);
      sb += fabsf(mt - mg);
      float d = ms - mt;
      sc += d * d;
    }
    // Whole wave belongs to one patch -> butterfly reduce.
#pragma unroll
    for (int off = 16; off > 0; off >>= 1) {
      sa += __shfl_xor(sa, off, 32);
      sb += __shfl_xor(sb, off, 32);
      sc += __shfl_xor(sc, off, 32);
    }
    if (lane == 0) {
      int p = nt * NT + tpatch; // patchify order: fpatch * nT + tpatch
      errS[(size_t)b * NPATCH + p] = sa;
      errT[(size_t)b * NPATCH + p] = sb;
      pl  [(size_t)b * NPATCH + p] = sc;
    }
  }
}

// ---------------------------------------------------------------------------
// Kernel 3: per-batch top-k by rank-counting + partial reductions.
// ---------------------------------------------------------------------------
__global__ __launch_bounds__(256)
void topk_batch(const float* __restrict__ errS, const float* __restrict__ errT,
                const float* __restrict__ pl, float* __restrict__ partials) {
  const int b = blockIdx.x;
  __shared__ float kgs[NPATCH];
  __shared__ float red[4][256];

  for (int p = threadIdx.x; p < NPATCH; p += 256)
    kgs[p] = (errS[(size_t)b * NPATCH + p] - errT[(size_t)b * NPATCH + p]) * (1.f / 256.f);
  __syncthreads();

  float selSum = 0.f, selCnt = 0.f, ksum = 0.f, pos = 0.f;
  for (int p = threadIdx.x; p < NPATCH; p += 256) {
    float v = kgs[p];
    int rank = 0;
    for (int j = 0; j < NPATCH; ++j) rank += (kgs[j] > v) ? 1 : 0;
    if (rank < KSEL) {
      selSum += pl[(size_t)b * NPATCH + p] * (1.f / 256.f);
      selCnt += 1.f;
    }
    ksum += v;
    pos  += (v > 0.f) ? 1.f : 0.f;
  }
  red[0][threadIdx.x] = selSum; red[1][threadIdx.x] = selCnt;
  red[2][threadIdx.x] = ksum;   red[3][threadIdx.x] = pos;
  __syncthreads();
  for (int s = 128; s > 0; s >>= 1) {
    if ((int)threadIdx.x < s)
#pragma unroll
      for (int q = 0; q < 4; ++q) red[q][threadIdx.x] += red[q][threadIdx.x + s];
    __syncthreads();
  }
  if (threadIdx.x == 0) {
    partials[b * 4 + 0] = red[0][0];
    partials[b * 4 + 1] = red[1][0];
    partials[b * 4 + 2] = red[2][0];
    partials[b * 4 + 3] = red[3][0];
  }
}

// ---------------------------------------------------------------------------
// Kernel 4: fold 16 batch partials into the 4 output scalars (one wave).
// ---------------------------------------------------------------------------
__global__ void finalize(const float* __restrict__ partials, float* __restrict__ out) {
  int t = threadIdx.x;
  float loss = 0.f, cnt = 0.f, ks = 0.f, pos = 0.f;
  if (t < BATCH) {
    float s = partials[t * 4 + 0];
    float c = partials[t * 4 + 1];
    loss = s / fmaxf(c, 1.f);
    cnt = c; ks = partials[t * 4 + 2]; pos = partials[t * 4 + 3];
  }
#pragma unroll
  for (int off = 16; off > 0; off >>= 1) {
    loss += __shfl_xor(loss, off, 32);
    cnt  += __shfl_xor(cnt,  off, 32);
    ks   += __shfl_xor(ks,   off, 32);
    pos  += __shfl_xor(pos,  off, 32);
  }
  if (t == 0) {
    const float invBP = 1.f / ((float)BATCH * (float)NPATCH);
    out[0] = loss / (float)BATCH; // loss
    out[1] = cnt * invBP;         // sel_ratio
    out[2] = ks * invBP;          // kgs_mean
    out[3] = pos * invBP;         // kgs_pos_ratio
  }
}

extern "C" void kernel_launch(void* const* d_in, const int* in_sizes, int n_in,
                              void* d_out, int out_size, void* d_ws, size_t ws_size,
                              hipStream_t stream) {
  (void)in_sizes; (void)n_in; (void)out_size; (void)ws_size;
  const float* xs = (const float*)d_in[0]; // student
  const float* xt = (const float*)d_in[1]; // teacher
  const float* xg = (const float*)d_in[2]; // target

  // Workspace layout (~27.7 MB total):
  char* ws = (char*)d_ws;
  _Float16* xh  = (_Float16*)ws;                      // 3*16*262144 halves
  _Float16* bre = xh + (size_t)3 * BATCH * L;
  _Float16* bim = bre + BASIS_HALVES;
  float* errS = (float*)(bim + BASIS_HALVES);
  float* errT = errS + (size_t)BATCH * NPATCH;
  float* pl   = errT + (size_t)BATCH * NPATCH;
  float* partials = pl + (size_t)BATCH * NPATCH;

  size_t n8 = (size_t)3 * BATCH * L / 8;
  to_half<<<(unsigned)((n8 + 255) / 256), 256, 0, stream>>>(xs, xt, xg, xh);
  build_basis<<<(BASIS_HALVES + 255) / 256, 256, 0, stream>>>(bre, bim);
  dim3 grid(NT, BATCH);
  stft_patch_stats<<<grid, 256, 0, stream>>>(xh, bre, bim, errS, errT, pl);
  topk_batch<<<BATCH, 256, 0, stream>>>(errS, errT, pl, partials);
  finalize<<<1, 32, 0, stream>>>(partials, (float*)d_out);
}
